// RotatedSingleStageDetectorAdv_807453852173
// MI455X (gfx1250) — matrix operand order repaired and checked by reading, not compiled, not run on hardware
//
#include <hip/hip_runtime.h>
#include <math.h>

#define B_   32
#define A_   3
#define C_   15
#define KSEL 2000
#define OUTN 6960
#define PI_F 3.14159265358979323846f
#define MAX_RATIO_F 4.135166556742356f   // |log(16/1000)|

typedef __attribute__((ext_vector_type(16))) _Float16 v16h;
typedef __attribute__((ext_vector_type(8)))  float    v8f;

__device__ __forceinline__ unsigned orderKey(float f) {
  unsigned u = __float_as_uint(f);
  return (u & 0x80000000u) ? ~u : (u | 0x80000000u);
}

// ---------------------------------------------------------------------------
// Stage 1: per anchor, max logit over 15 classes -> orderable uint key.
// ---------------------------------------------------------------------------
__global__ void k_maxkey(const float* __restrict__ cls, unsigned* __restrict__ keys,
                         int HW, int total4) {
  int t = blockIdx.x * blockDim.x + threadIdx.x;
  if (t >= total4) return;
  int perB = HW >> 2;
  int b    = t / perB;
  int hw4  = (t - b * perB) << 2;
  const float* base = cls + (size_t)b * (A_ * C_) * HW + hw4;
  __builtin_prefetch(base, 0, 3);  // gfx1250 global_prefetch_b8
  unsigned out[12];
  #pragma unroll
  for (int a = 0; a < A_; ++a) {
    const float* pa = base + (size_t)(a * C_) * HW;
    float4 m = *(const float4*)pa;
    for (int c = 1; c < C_; ++c) {
      float4 v = *(const float4*)(pa + (size_t)c * HW);
      m.x = fmaxf(m.x, v.x); m.y = fmaxf(m.y, v.y);
      m.z = fmaxf(m.z, v.z); m.w = fmaxf(m.w, v.w);
    }
    out[0 * A_ + a] = orderKey(m.x);
    out[1 * A_ + a] = orderKey(m.y);
    out[2 * A_ + a] = orderKey(m.z);
    out[3 * A_ + a] = orderKey(m.w);
  }
  unsigned* kp = keys + (size_t)b * ((size_t)HW * A_) + (size_t)hw4 * A_;
  ((uint4*)kp)[0] = make_uint4(out[0], out[1], out[2],  out[3]);
  ((uint4*)kp)[1] = make_uint4(out[4], out[5], out[6],  out[7]);
  ((uint4*)kp)[2] = make_uint4(out[8], out[9], out[10], out[11]);
}

// ---------------------------------------------------------------------------
// WMMA-based descending cumulative (suffix) sum over the 256-bin histogram +
// digit pick.  Runs on wave 0 (full 32-lane EXEC).  hist counts < 2^16 are
// split into byte planes (f16-exact); all accumulation is in f32 (exact).
//   H[i][j] = hist[16*j + i]
//   cum[16j+i] = (Ut·H)[i][j] + (Ones·(H·M))[i][j]
// with Ut[i][k]=1 iff k>=i,  M[k][j]=1 iff k>j,  Ones = all-ones.
// Intermediate T=H·M (<=3825) is re-split into lo/hi bytes; hi plane is
// scaled via an all-256 A matrix (256 is f16-exact).
// ---------------------------------------------------------------------------
__device__ __forceinline__ void wmma_pick_digit(const unsigned* hist, int shift,
                                                unsigned* sh_prefix,
                                                unsigned* sh_remaining,
                                                float* tmpT, int lane) {
  const unsigned prefix = *sh_prefix;
  const unsigned rem    = *sh_remaining;
  const int hf = lane >> 4;     // lane half: 0 or 1
  const int lm = lane & 15;
  const _Float16 h1   = (_Float16)1.0f;
  const _Float16 h256 = (_Float16)256.0f;

  // A operands (16x32, zero-padded for K>=16): element e<8 <-> K = 8*hf + e
  v16h A_ut = {}, A_ones = {}, A_256 = {}, A_h0 = {}, A_h1 = {};
  #pragma unroll
  for (int e = 0; e < 8; ++e) {
    int K = 8 * hf + e;
    A_ut[e]   = (K >= lm) ? h1 : (_Float16)0.0f;   // row M = lm
    A_ones[e] = h1;
    A_256[e]  = h256;
    unsigned hv = hist[K * 16 + lm];               // H[lm][K]
    A_h0[e] = (_Float16)(float)(hv & 255u);
    A_h1[e] = (_Float16)(float)(hv >> 8);
  }
  // B operands (32x16, rows K>=16 zero): element e<8 <-> K = 8*hf + e, N = lm
  v16h B_h0 = {}, B_h1 = {}, B_m = {};
  {
    #pragma unroll
    for (int e = 0; e < 8; ++e) {
      int K = 8 * hf + e;
      unsigned hv = hist[lm * 16 + K];             // H[K][lm] -> B[K][N=lm]
      B_h0[e] = (_Float16)(float)(hv & 255u);
      B_h1[e] = (_Float16)(float)(hv >> 8);
      B_m[e]  = (K > lm) ? h1 : (_Float16)0.0f;    // M[k=K][j=lm]
    }
  }
  v8f z = {};
  v8f S0 = __builtin_amdgcn_wmma_f32_16x16x32_f16(false, A_ut, false, B_h0, (short)0, z, false, false);
  v8f S1 = __builtin_amdgcn_wmma_f32_16x16x32_f16(false, A_ut, false, B_h1, (short)0, z, false, false);
  v8f T0 = __builtin_amdgcn_wmma_f32_16x16x32_f16(false, A_h0, false, B_m, (short)0, z, false, false);
  v8f T1 = __builtin_amdgcn_wmma_f32_16x16x32_f16(false, A_h1, false, B_m, (short)0, z, false, false);

  // D layout: VGPR r, lanes 0-15 -> (M=r, N=lane), lanes 16-31 -> (M=8+r)
  #pragma unroll
  for (int r = 0; r < 8; ++r) {
    int i = r + 8 * hf;
    tmpT[i * 16 + lm]       = T0[r];
    tmpT[256 + i * 16 + lm] = T1[r];
  }
  // rebuild T planes as B operands (byte-split: values <= 3825)
  v16h B_t0lo = {}, B_t0hi = {}, B_t1lo = {}, B_t1hi = {};
  {
    #pragma unroll
    for (int e = 0; e < 8; ++e) {
      int K = 8 * hf + e;
      unsigned t0 = (unsigned)(tmpT[K * 16 + lm] + 0.5f);
      unsigned t1 = (unsigned)(tmpT[256 + K * 16 + lm] + 0.5f);
      B_t0lo[e] = (_Float16)(float)(t0 & 255u);
      B_t0hi[e] = (_Float16)(float)(t0 >> 8);
      B_t1lo[e] = (_Float16)(float)(t1 & 255u);
      B_t1hi[e] = (_Float16)(float)(t1 >> 8);
    }
  }
  v8f C0   = __builtin_amdgcn_wmma_f32_16x16x32_f16(false, A_256,  false, B_t0hi, (short)0, S0, false, false);
  v8f CUM0 = __builtin_amdgcn_wmma_f32_16x16x32_f16(false, A_ones, false, B_t0lo, (short)0, C0, false, false);
  v8f C1   = __builtin_amdgcn_wmma_f32_16x16x32_f16(false, A_256,  false, B_t1hi, (short)0, S1, false, false);
  v8f CUM1 = __builtin_amdgcn_wmma_f32_16x16x32_f16(false, A_ones, false, B_t1lo, (short)0, C1, false, false);

  // each lane owns 8 bins: bin = 16*N + M = 16*lm + r + 8*hf
  #pragma unroll
  for (int r = 0; r < 8; ++r) {
    float cf = CUM0[r] + 256.0f * CUM1[r];
    unsigned cu = (unsigned)(cf + 0.5f);           // exact integer suffix sum
    int bin = 16 * lm + r + 8 * hf;
    unsigned above = cu - hist[bin];               // count of digits > bin
    if (cu >= rem && above < rem) {                // unique crossing digit
      *sh_prefix    = prefix | ((unsigned)bin << shift);
      *sh_remaining = rem - above;
    }
  }
}

// ---------------------------------------------------------------------------
// Stage 2: exact top-2000 per (batch, level), jax.lax.top_k ordering.
// Keys staged global -> LDS once via async DMA; all passes then hit LDS.
// ---------------------------------------------------------------------------
__global__ __launch_bounds__(1024) void k_select(const unsigned* __restrict__ ws_keys,
                                                 unsigned* __restrict__ topk) {
  const int lvlN[3]   = {49152, 12288, 3072};
  const int lvlOff[3] = {0, B_ * 49152, B_ * 49152 + B_ * 12288};
  int bi  = blockIdx.x;
  int b   = bi / 3;
  int lvl = bi - b * 3;
  int N   = lvlN[lvl];
  const unsigned* keys = ws_keys + lvlOff[lvl] + (size_t)b * N;
  int tid = threadIdx.x;

  extern __shared__ unsigned lkeys[];          // N keys (dyn-LDS, <=192KB)
  __shared__ unsigned hist[256];
  __shared__ unsigned sh_prefix, sh_remaining, sh_c1, sh_carry;
  __shared__ unsigned scanbuf[1024];
  __shared__ float    tmpT[512];
  __shared__ unsigned long long sortbuf[2048];

  // ---- stage keys into LDS with gfx1250 async DMA (b128 per issue) ----
  for (int i4 = tid; i4 < (N >> 2); i4 += 1024) {
    unsigned           lds_b = (unsigned)(size_t)(&lkeys[i4 << 2]);
    unsigned long long ga    = (unsigned long long)(size_t)(keys + (i4 << 2));
    asm volatile("global_load_async_to_lds_b128 %0, %1, off"
                 :: "v"(lds_b), "v"(ga) : "memory");
  }
  asm volatile("s_wait_asynccnt 0x0" ::: "memory");
  if (tid == 0) { sh_prefix = 0u; sh_remaining = KSEL; }
  __syncthreads();

  // ---- radix select, 8-bit digits from MSB ----
  for (int pass = 0; pass < 4; ++pass) {
    int shift = 24 - 8 * pass;
    unsigned maskHi = (pass == 0) ? 0u : (0xFFFFFFFFu << (shift + 8));
    unsigned prefix = sh_prefix;
    if (tid < 256) hist[tid] = 0u;
    __syncthreads();
    for (int i = tid; i < N; i += 1024) {
      unsigned k = lkeys[i];
      if ((k & maskHi) == prefix)
        atomicAdd(&hist[(k >> shift) & 255u], 1u);
    }
    __syncthreads();
    if (tid < 32)   // wave 0, full EXEC: WMMA suffix-scan + digit pick
      wmma_pick_digit(hist, shift, &sh_prefix, &sh_remaining, tmpT, tid);
    __syncthreads();
  }
  const unsigned T = sh_prefix;        // K-th largest key value
  const unsigned r = sh_remaining;     // #ties at T to keep (smallest indices)
  const unsigned c1base = KSEL - r;    // #keys strictly greater than T

  // ---- fill sort buffer ----
  for (int i = tid; i < 2048; i += 1024) sortbuf[i] = ~0ull;
  if (tid == 0) { sh_c1 = 0u; sh_carry = 0u; }
  __syncthreads();
  for (int i = tid; i < N; i += 1024) {
    unsigned k = lkeys[i];
    if (k > T) {
      unsigned p = atomicAdd(&sh_c1, 1u);
      if (p < c1base)
        sortbuf[p] = ((unsigned long long)(~k) << 32) | (unsigned)i;
    }
  }
  __syncthreads();

  // ---- ties: r smallest indices via stable chunked block prefix scan ----
  const unsigned long long tieKey = ((unsigned long long)(~T) << 32);
  for (int base_i = 0; base_i < N; base_i += 1024) {
    if (sh_carry >= r) break;                 // uniform
    unsigned carry = sh_carry;
    __syncthreads();
    int i = base_i + tid;
    unsigned flag = (i < N && lkeys[i] == T) ? 1u : 0u;
    scanbuf[tid] = flag;
    __syncthreads();
    for (int off = 1; off < 1024; off <<= 1) {
      unsigned v = scanbuf[tid];
      if (tid >= off) v += scanbuf[tid - off];
      __syncthreads();
      scanbuf[tid] = v;
      __syncthreads();
    }
    if (flag) {
      unsigned rank = carry + scanbuf[tid] - 1u;
      if (rank < r) sortbuf[c1base + rank] = tieKey | (unsigned)i;
    }
    unsigned tot = scanbuf[1023];
    __syncthreads();
    if (tid == 0) sh_carry = carry + tot;
    __syncthreads();
  }
  __syncthreads();

  // ---- bitonic sort ascending over 2048 composites ----
  for (int k = 2; k <= 2048; k <<= 1) {
    for (int j = k >> 1; j > 0; j >>= 1) {
      for (int i = tid; i < 2048; i += 1024) {
        int ij = i ^ j;
        if (ij > i) {
          unsigned long long va = sortbuf[i], vb = sortbuf[ij];
          bool up = ((i & k) == 0);
          if ((va > vb) == up) { sortbuf[i] = vb; sortbuf[ij] = va; }
        }
      }
      __syncthreads();
    }
  }

  for (int s = tid; s < KSEL; s += 1024)
    topk[(size_t)bi * KSEL + s] = (unsigned)(sortbuf[s] & 0xFFFFFFFFu);
}

// ---------------------------------------------------------------------------
// Stage 3: gather + sigmoid + rotated decode (pa = 0 => cos=1, sin=0).
// ---------------------------------------------------------------------------
__global__ void k_decode(const float* __restrict__ cls0, const float* __restrict__ cls1,
                         const float* __restrict__ cls2, const float* __restrict__ cls3,
                         const float* __restrict__ cls4,
                         const float* __restrict__ box0, const float* __restrict__ box1,
                         const float* __restrict__ box2, const float* __restrict__ box3,
                         const float* __restrict__ box4,
                         const unsigned* __restrict__ topk,
                         float* __restrict__ out_bbox, float* __restrict__ out_score) {
  int t = blockIdx.x * blockDim.x + threadIdx.x;
  if (t >= B_ * OUTN) return;
  int b = t / OUTN;
  int slot = t - b * OUTN;
  int lvl, off;
  if      (slot < 2000) { lvl = 0; off = 0;    }
  else if (slot < 4000) { lvl = 1; off = 2000; }
  else if (slot < 6000) { lvl = 2; off = 4000; }
  else if (slot < 6768) { lvl = 3; off = 6000; }
  else                  { lvl = 4; off = 6768; }

  const int   Wl[5]      = {128, 64, 32, 16, 8};
  const int   strideL[5] = {8, 16, 32, 64, 128};
  const float hrL[3]     = {0.70710678118654752f, 1.0f, 1.41421356237309505f};

  const float* cls; const float* box;
  switch (lvl) {
    case 0:  cls = cls0; box = box0; break;
    case 1:  cls = cls1; box = box1; break;
    case 2:  cls = cls2; box = box2; break;
    case 3:  cls = cls3; box = box3; break;
    default: cls = cls4; box = box4; break;
  }
  int W  = Wl[lvl];
  int HW = W * W;
  int n;
  if (lvl < 3) n = (int)topk[(size_t)(b * 3 + lvl) * KSEL + (slot - off)];
  else         n = slot - off;
  int a  = n % A_;
  int hw = n / A_;
  int w  = hw % W;
  int h  = hw / W;

  const float* cb = cls + ((size_t)b * (A_ * C_) + (size_t)(a * C_)) * HW + hw;
  float* so = out_score + (size_t)t * 16;
  #pragma unroll
  for (int c = 0; c < C_; ++c) {
    float x = cb[(size_t)c * HW];
    so[c] = 1.0f / (1.0f + expf(-x));
  }
  so[15] = 0.0f;

  const float* db = box + ((size_t)b * (A_ * 5) + (size_t)(a * 5)) * HW + hw;
  float dx = db[0];
  float dy = db[(size_t)HW];
  float dw = db[(size_t)2 * HW];
  float dh = db[(size_t)3 * HW];
  float da = db[(size_t)4 * HW];
  dw = fminf(fmaxf(dw, -MAX_RATIO_F), MAX_RATIO_F);
  dh = fminf(fmaxf(dh, -MAX_RATIO_F), MAX_RATIO_F);

  float stride = (float)strideL[lvl];
  float hr = hrL[a];
  float pw = stride * 8.0f / hr;
  float ph = stride * 8.0f * hr;
  float px = ((float)w + 0.5f) * stride;
  float py = ((float)h + 0.5f) * stride;

  float gx = px + pw * dx;
  float gy = py + ph * dy;
  float gw = pw * expf(dw);
  float gh = ph * expf(dh);
  float ang = da + PI_F * 0.5f;
  ang = ang - floorf(ang / PI_F) * PI_F;   // python-style mod -> [0, pi)
  float ga = ang - PI_F * 0.5f;

  float* bo = out_bbox + (size_t)t * 5;
  bo[0] = gx; bo[1] = gy; bo[2] = gw; bo[3] = gh; bo[4] = ga;
}

// ---------------------------------------------------------------------------
extern "C" void kernel_launch(void* const* d_in, const int* in_sizes, int n_in,
                              void* d_out, int out_size, void* d_ws, size_t ws_size,
                              hipStream_t stream) {
  (void)in_sizes; (void)n_in; (void)out_size; (void)ws_size;
  // setup_inputs() dict order is interleaved: cls_s0, box_s0, cls_s1, box_s1, ...
  const float* cls[5]; const float* box[5];
  for (int i = 0; i < 5; ++i) {
    cls[i] = (const float*)d_in[2 * i];
    box[i] = (const float*)d_in[2 * i + 1];
  }
  unsigned* ws_keys = (unsigned*)d_ws;
  size_t keyCount = (size_t)B_ * (49152 + 12288 + 3072);
  unsigned* topk = ws_keys + keyCount;     // 32*3*2000 uints

  float* out_bbox  = (float*)d_out;
  float* out_score = out_bbox + (size_t)B_ * OUTN * 5;

  const int HWs[3] = {128 * 128, 64 * 64, 32 * 32};
  const int keyOff[3] = {0, B_ * 49152, B_ * 49152 + B_ * 12288};
  for (int l = 0; l < 3; ++l) {
    int HW = HWs[l];
    int total4 = B_ * (HW >> 2);
    int blk = 256;
    int grid = (total4 + blk - 1) / blk;
    k_maxkey<<<grid, blk, 0, stream>>>(cls[l], ws_keys + keyOff[l], HW, total4);
  }
  // dynamic LDS = level-0 key slab (192 KB; MI455X WGP has 320 KB)
  k_select<<<B_ * 3, 1024, 49152 * sizeof(unsigned), stream>>>(ws_keys, topk);
  {
    int tot = B_ * OUTN;
    int blk = 256;
    k_decode<<<(tot + blk - 1) / blk, blk, 0, stream>>>(
        cls[0], cls[1], cls[2], cls[3], cls[4],
        box[0], box[1], box[2], box[3], box[4],
        topk, out_bbox, out_score);
  }
}
